// HeteroGCN_36129264894743
// MI455X (gfx1250) — matrix operand order repaired and checked by reading, not compiled, not run on hardware
//
#include <hip/hip_runtime.h>
#include <hip/hip_bf16.h>

// ---------------- problem constants ----------------
#define N_AC 100000
#define N_WD 20000
#define D_IN 128
#define H1   256
#define H2   128
#define LINH 64
#define OUTD 10
#define E_TIC 1600000L
#define E_W   320000L
#define E_REL 800000L

// ---------------- workspace layout (float offsets) ----------------
// tight aliasing: bufA = agg1_tic -> y_tic -> ha2 ; bufB = agg1_rel -> y_w ;
// bufC = agg1_w -> y_rel ; agg2_tic -> h1 ; agg2_rel -> h2
#define OFF_DEG_TIC 0L
#define OFF_DEG_REL 100000L
#define OFF_DEG_W   200000L
#define OFF_BUFA    220000L                       // 12,800,000
#define OFF_BUFB    13020000L                     // 12,800,000
#define OFF_BUFC    25820000L                     //  2,560,000
#define OFF_HA      28380000L                     // 25,600,000
#define OFF_HW      53980000L                     //  5,120,000
#define OFF_WC1     59100000L                     //     32,768
#define OFF_WC2     59132768L                     //     32,768
#define OFF_AGG2T   59165536L                     // 12,800,000
#define OFF_AGG2R   71965536L                     // 12,800,000
#define OFF_AGG2W   84765536L                     //  2,560,000
#define WS_FLOATS   87325536L

typedef __attribute__((ext_vector_type(16))) __bf16 v16bf;
typedef __attribute__((ext_vector_type(8)))  __bf16 v8bf;
typedef __attribute__((ext_vector_type(8)))  float  v8f;

// ---------------- small utility kernels ----------------
__global__ __launch_bounds__(256) void zero_f32v4(float4* __restrict__ p, long n4) {
    long i = (long)blockIdx.x * 256 + threadIdx.x;
    if (i < n4) p[i] = make_float4(0.f, 0.f, 0.f, 0.f);
}

__global__ __launch_bounds__(256) void vec_add_f32(const float* __restrict__ a,
                                                   const float* __restrict__ b,
                                                   float* __restrict__ c, int n) {
    int i = blockIdx.x * 256 + threadIdx.x;
    if (i < n) c[i] = a[i] + b[i];
}

__global__ __launch_bounds__(256) void deg_count(const int* __restrict__ dst,
                                                 float* __restrict__ deg, long E) {
    long e = (long)blockIdx.x * 256 + threadIdx.x;
    if (e < E) atomicAdd(&deg[dst[e]], 1.0f);
}

// msg = w_e * xsrc[src[e]] ; agg[dst[e]] += msg   (feature dim fixed at 128)
__global__ __launch_bounds__(256) void scatter_add128(const float* __restrict__ xsrc,
                                                      const int* __restrict__ src,
                                                      const int* __restrict__ dst,
                                                      const float* __restrict__ w,
                                                      float* __restrict__ agg, long E) {
    long idx = (long)blockIdx.x * 256 + threadIdx.x;
    long e = idx >> 7;
    int  d = (int)(idx & 127);
    if (e < E) {
        int   s  = src[e];
        int   t  = dst[e];
        float we = w[e];
        atomicAdd(&agg[(long)t * 128 + d], we * xsrc[(long)s * 128 + d]);
    }
}

// agg[r,:] /= max(deg[r],1)   (D = 128)
__global__ __launch_bounds__(256) void rownorm128(float* __restrict__ agg,
                                                  const float* __restrict__ deg, long M) {
    long idx = (long)blockIdx.x * 256 + threadIdx.x;
    if (idx < M * 128L) {
        long r = idx >> 7;
        agg[idx] = agg[idx] / fmaxf(deg[r], 1.0f);
    }
}

// ---------------- bf16 WMMA GEMM ----------------
// out[M,N] = act( alpha * ( sum_p A_p @ W_p  + add0 + add1 + bias0 + bias1 ) )
// A_p : [M,K] f32 row-major, W_p : [K,N] f32 row-major, add* : [M,N] f32, bias* : [N]
// Block: 256 threads = 8 waves (wave32). Block tile 128M x 64N; each wave owns a
// 32M x 32N sub-tile -> 2 A-frags x 2 B-frags = 4 v_wmma_f32_16x16x32_bf16 per
// K-step. Staging is branchless: OOB rows/cols are clamped (they only feed C
// entries that are never stored). f32->bf16 conversion happens on the LDS path.
// NFAST (compile-time): the 64-col W panel is fully in-bounds and 16B aligned.
template <int NPAIRS, bool RELU, bool NFAST>
__global__ __launch_bounds__(256) void gemm_bf16_wmma(
    const float* __restrict__ A0, const float* __restrict__ W0,
    const float* __restrict__ A1, const float* __restrict__ W1,
    const float* __restrict__ A2, const float* __restrict__ W2,
    const float* __restrict__ bias0, const float* __restrict__ bias1,
    const float* __restrict__ add0,  const float* __restrict__ add1,
    float* __restrict__ out, int M, int N, int K, float alpha)
{
    // A staged row-major [128][stride 40]; W staged transposed [n][k] [64][stride 40]
    __shared__ __align__(16) __bf16 As[128 * 40];
    __shared__ __align__(16) __bf16 Bs[64 * 40];

    const int tid  = threadIdx.x;
    const int lane = tid & 31;
    const int wid  = tid >> 5;
    const int wm   = wid >> 1;   // 0..3 : wave row block (32 rows each)
    const int wn   = wid & 1;    // 0..1 : wave col block (32 cols each)
    const int bm   = blockIdx.x;
    const int bn   = blockIdx.y;

    v8f cc[2][2];
    cc[0][0] = (v8f){}; cc[0][1] = (v8f){};
    cc[1][0] = (v8f){}; cc[1][1] = (v8f){};

#pragma unroll
    for (int p = 0; p < NPAIRS; ++p) {
        // NPAIRS is a template constant and the loop is fully unrolled, so these
        // resolve to direct kernarg pointers (keeps global address-space inference
        // -> global_load_b128, not flat_load).
        const float* __restrict__ A  = (p == 0) ? A0 : (p == 1) ? A1 : A2;
        const float* __restrict__ Wt = (p == 0) ? W0 : (p == 1) ? W1 : W2;
        for (int kk = 0; kk < K; kk += 32) {
            // ---- stage A tile: 128 rows x 32 k (two 64-row passes) ----
            {
                int row = tid >> 2;
                int kc  = (tid & 3) * 8;
#pragma unroll
                for (int r2 = 0; r2 < 128; r2 += 64) {
                    long gm = (long)bm * 128 + row + r2;
                    if (gm > M - 1) gm = M - 1;   // clamp: rows >= M never stored
                    const float4* g = (const float4*)(A + gm * (long)K + kk + kc);
                    float4 f0 = g[0];
                    float4 f1 = g[1];
                    v8bf pk;
                    pk[0] = (__bf16)f0.x; pk[1] = (__bf16)f0.y;
                    pk[2] = (__bf16)f0.z; pk[3] = (__bf16)f0.w;
                    pk[4] = (__bf16)f1.x; pk[5] = (__bf16)f1.y;
                    pk[6] = (__bf16)f1.z; pk[7] = (__bf16)f1.w;
                    *(v8bf*)&As[(row + r2) * 40 + kc] = pk;
                }
            }
            // ---- stage W tile transposed: Bs[n][k], 32 k x 64 n ----
            {
                int  k   = tid >> 3;
                int  nc  = (tid & 7) * 8;
                long gk  = kk + k;
                int  gn0 = bn * 64 + nc;
                float f[8];
                if (NFAST) {
                    const float4* g = (const float4*)(Wt + gk * (long)N + gn0);
                    float4 f0 = g[0];
                    float4 f1 = g[1];
                    f[0] = f0.x; f[1] = f0.y; f[2] = f0.z; f[3] = f0.w;
                    f[4] = f1.x; f[5] = f1.y; f[6] = f1.z; f[7] = f1.w;
                } else {
#pragma unroll
                    for (int i = 0; i < 8; ++i) {
                        int gn = gn0 + i;
                        if (gn > N - 1) gn = N - 1;  // clamp: cols >= N never stored
                        f[i] = Wt[gk * (long)N + gn];
                    }
                }
#pragma unroll
                for (int i = 0; i < 8; ++i)
                    Bs[(nc + i) * 40 + k] = (__bf16)f[i];
            }
            __syncthreads();
            // ---- assemble fragments per ISA VGPR layout, issue 4 WMMAs ----
            {
                // A 16x32 bf16: lanes 0-15 row=lane, K[0:8)+K[16:24);
                //               lanes16-31 same row, K[8:16)+K[24:32)
                int khalf = lane >> 4;
                v16bf a[2], b[2];
#pragma unroll
                for (int sm = 0; sm < 2; ++sm) {
                    int arow = wm * 32 + sm * 16 + (lane & 15);
                    const v8bf* ap = (const v8bf*)&As[arow * 40 + khalf * 8];
                    a[sm] = __builtin_shufflevector(ap[0], ap[2],
                        0, 1, 2, 3, 4, 5, 6, 7, 8, 9, 10, 11, 12, 13, 14, 15);
                }
                // B 32x16 bf16: lane%16 = column, lanes 0-15 K[0:8)+K[16:24), lanes 16-31 K[8:16)+K[24:32)
                int kb = khalf * 8;
#pragma unroll
                for (int sn = 0; sn < 2; ++sn) {
                    int ncol = wn * 32 + sn * 16 + (lane & 15);
                    const v8bf* bp = (const v8bf*)&Bs[ncol * 40 + kb];
                    b[sn] = __builtin_shufflevector(bp[0], bp[2],
                        0, 1, 2, 3, 4, 5, 6, 7, 8, 9, 10, 11, 12, 13, 14, 15);
                }
#pragma unroll
                for (int sm = 0; sm < 2; ++sm)
#pragma unroll
                    for (int sn = 0; sn < 2; ++sn)
                        cc[sm][sn] = __builtin_amdgcn_wmma_f32_16x16x32_bf16(
                            false, a[sm], false, b[sn], (short)0, cc[sm][sn],
                            false, false);
            }
            __syncthreads();
        }
    }

    // ---- epilogue: C layout lanes0-15 M=r, lanes16-31 M=r+8; N = lane%16 ----
#pragma unroll
    for (int sm = 0; sm < 2; ++sm) {
        int mBase = bm * 128 + wm * 32 + sm * 16 + 8 * (lane >> 4);
#pragma unroll
        for (int sn = 0; sn < 2; ++sn) {
            int n = bn * 64 + wn * 32 + sn * 16 + (lane & 15);
            if (n < N) {
                float bsum = 0.f;
                if (bias0) bsum += bias0[n];
                if (bias1) bsum += bias1[n];
                v8f c = cc[sm][sn];
#pragma unroll
                for (int r = 0; r < 8; ++r) {
                    long m = mBase + r;
                    if (m < M) {
                        float v = c[r] + bsum;
                        if (add0) v += add0[m * (long)N + n];
                        if (add1) v += add1[m * (long)N + n];
                        v *= alpha;
                        if (RELU) v = fmaxf(v, 0.f);
                        out[m * (long)N + n] = v;
                    }
                }
            }
        }
    }
}

// ---------------- host orchestration ----------------
static inline long cdiv(long a, long b) { return (a + b - 1) / b; }

extern "C" void kernel_launch(void* const* d_in, const int* in_sizes, int n_in,
                              void* d_out, int out_size, void* d_ws, size_t ws_size,
                              hipStream_t stream)
{
    (void)in_sizes; (void)n_in; (void)out_size;
    if (ws_size < (size_t)WS_FLOATS * sizeof(float)) return;  // workspace too small

    const float* x_ac   = (const float*)d_in[0];
    const float* x_wd   = (const float*)d_in[1];
    const float* w_tic  = (const float*)d_in[2];
    const float* w_w    = (const float*)d_in[3];
    const float* w_rel  = (const float*)d_in[4];
    const int*   s_tic  = (const int*)d_in[5];
    const int*   t_tic  = (const int*)d_in[6];
    const int*   s_w    = (const int*)d_in[7];
    const int*   t_w    = (const int*)d_in[8];
    const int*   s_rel  = (const int*)d_in[9];
    const int*   t_rel  = (const int*)d_in[10];
    const float *Ws1_tic=(const float*)d_in[11], *Wn1_tic=(const float*)d_in[12], *b1_tic=(const float*)d_in[13];
    const float *Ws1_w  =(const float*)d_in[14], *Wn1_w  =(const float*)d_in[15], *b1_w  =(const float*)d_in[16];
    const float *Ws1_rel=(const float*)d_in[17], *Wn1_rel=(const float*)d_in[18], *b1_rel=(const float*)d_in[19];
    const float *Ws2_tic=(const float*)d_in[20], *Wn2_tic=(const float*)d_in[21], *b2_tic=(const float*)d_in[22];
    const float *Ws2_w  =(const float*)d_in[23], *Wn2_w  =(const float*)d_in[24], *b2_w  =(const float*)d_in[25];
    const float *Ws2_rel=(const float*)d_in[26], *Wn2_rel=(const float*)d_in[27], *b2_rel=(const float*)d_in[28];
    const float *W_l1=(const float*)d_in[29], *b_l1=(const float*)d_in[30];
    const float *W_l2=(const float*)d_in[31], *b_l2=(const float*)d_in[32];
    const float *W_l3=(const float*)d_in[33], *b_l3=(const float*)d_in[34];

    float* W = (float*)d_ws;
    float* deg_tic = W + OFF_DEG_TIC;
    float* deg_rel = W + OFF_DEG_REL;
    float* deg_w   = W + OFF_DEG_W;
    float* bufA    = W + OFF_BUFA;   // agg1_tic -> y_tic -> ha2
    float* bufB    = W + OFF_BUFB;   // agg1_rel -> y_w
    float* bufC    = W + OFF_BUFC;   // agg1_w   -> y_rel
    float* ha      = W + OFF_HA;
    float* hw      = W + OFF_HW;
    float* Wc1     = W + OFF_WC1;
    float* Wc2     = W + OFF_WC2;
    float* agg2t   = W + OFF_AGG2T;  // -> h1
    float* agg2r   = W + OFF_AGG2R;  // -> h2
    float* agg2w   = W + OFF_AGG2W;

    float* out_head = (float*)d_out;                     // [N_AC, 10]
    float* out_hw2  = (float*)d_out + (long)N_AC * OUTD; // [N_W, 128]

    const long MT_AC = cdiv(N_AC, 128);  // 782
    const long MT_W  = cdiv(N_WD, 128);  // 157

    // 0) zero deg + layer-1 aggregates, and layer-2 aggregates
    {
        long n4 = OFF_HA / 4;  // [0, OFF_HA) covers degs + agg1 buffers
        zero_f32v4<<<cdiv(n4, 256), 256, 0, stream>>>((float4*)W, n4);
        long n4b = (WS_FLOATS - OFF_AGG2T) / 4;  // agg2_tic..agg2_w contiguous
        zero_f32v4<<<cdiv(n4b, 256), 256, 0, stream>>>((float4*)agg2t, n4b);
    }

    // 1) combined self-weights: Wc1 = Ws1_tic + Ws1_rel, Wc2 = Ws2_tic + Ws2_rel
    vec_add_f32<<<cdiv(D_IN * H1, 256), 256, 0, stream>>>(Ws1_tic, Ws1_rel, Wc1, D_IN * H1);
    vec_add_f32<<<cdiv(H1 * H2, 256), 256, 0, stream>>>(Ws2_tic, Ws2_rel, Wc2, H1 * H2);

    // 2) degrees per edge type
    deg_count<<<cdiv(E_TIC, 256), 256, 0, stream>>>(t_tic, deg_tic, E_TIC);
    deg_count<<<cdiv(E_REL, 256), 256, 0, stream>>>(t_rel, deg_rel, E_REL);
    deg_count<<<cdiv(E_W,   256), 256, 0, stream>>>(t_w,   deg_w,   E_W);

    // 3) layer-1 scatter (din = 128) then normalize
    scatter_add128<<<cdiv(E_TIC * 128, 256), 256, 0, stream>>>(x_ac, s_tic, t_tic, w_tic, bufA, E_TIC);
    scatter_add128<<<cdiv(E_REL * 128, 256), 256, 0, stream>>>(x_wd, s_rel, t_rel, w_rel, bufB, E_REL);
    scatter_add128<<<cdiv(E_W   * 128, 256), 256, 0, stream>>>(x_wd, s_w,   t_w,   w_w,   bufC, E_W);
    rownorm128<<<cdiv((long)N_AC * 128, 256), 256, 0, stream>>>(bufA, deg_tic, N_AC);
    rownorm128<<<cdiv((long)N_AC * 128, 256), 256, 0, stream>>>(bufB, deg_rel, N_AC);
    rownorm128<<<cdiv((long)N_WD * 128, 256), 256, 0, stream>>>(bufC, deg_w,   N_WD);

    // 4) layer-1 GEMMs (WMMA):
    // ha = relu(0.5*(x_ac@Wc1 + n_tic@Wn1_tic + n_rel@Wn1_rel + b1_tic + b1_rel))
    gemm_bf16_wmma<3, true, true><<<dim3(MT_AC, cdiv(H1, 64)), 256, 0, stream>>>(
        x_ac, Wc1, bufA, Wn1_tic, bufB, Wn1_rel,
        b1_tic, b1_rel, nullptr, nullptr, ha, N_AC, H1, D_IN, 0.5f);
    // hw = relu(x_wd@Ws1_w + n_w@Wn1_w + b1_w)
    gemm_bf16_wmma<2, true, true><<<dim3(MT_W, cdiv(H1, 64)), 256, 0, stream>>>(
        x_wd, Ws1_w, bufC, Wn1_w, nullptr, nullptr,
        b1_w, nullptr, nullptr, nullptr, hw, N_WD, H1, D_IN, 1.0f);

    // 5) layer-2: project first (dout = 128 < din = 256), then scatter projected feats
    gemm_bf16_wmma<1, false, true><<<dim3(MT_AC, cdiv(H2, 64)), 256, 0, stream>>>(
        ha, Wn2_tic, nullptr, nullptr, nullptr, nullptr,
        nullptr, nullptr, nullptr, nullptr, bufA, N_AC, H2, H1, 1.0f);  // y_tic
    gemm_bf16_wmma<1, false, true><<<dim3(MT_W, cdiv(H2, 64)), 256, 0, stream>>>(
        hw, Wn2_rel, nullptr, nullptr, nullptr, nullptr,
        nullptr, nullptr, nullptr, nullptr, bufC, N_WD, H2, H1, 1.0f);  // y_rel
    gemm_bf16_wmma<1, false, true><<<dim3(MT_W, cdiv(H2, 64)), 256, 0, stream>>>(
        hw, Wn2_w, nullptr, nullptr, nullptr, nullptr,
        nullptr, nullptr, nullptr, nullptr, bufB, N_WD, H2, H1, 1.0f);  // y_w

    scatter_add128<<<cdiv(E_TIC * 128, 256), 256, 0, stream>>>(bufA, s_tic, t_tic, w_tic, agg2t, E_TIC);
    scatter_add128<<<cdiv(E_REL * 128, 256), 256, 0, stream>>>(bufC, s_rel, t_rel, w_rel, agg2r, E_REL);
    scatter_add128<<<cdiv(E_W   * 128, 256), 256, 0, stream>>>(bufB, s_w,   t_w,   w_w,   agg2w, E_W);
    rownorm128<<<cdiv((long)N_AC * 128, 256), 256, 0, stream>>>(agg2t, deg_tic, N_AC);
    rownorm128<<<cdiv((long)N_AC * 128, 256), 256, 0, stream>>>(agg2r, deg_rel, N_AC);
    rownorm128<<<cdiv((long)N_WD * 128, 256), 256, 0, stream>>>(agg2w, deg_w,   N_WD);

    // 6) layer-2 combine GEMMs:
    // ha2 = 0.5*(ha@Wc2 + agg2t + agg2r + b2_tic + b2_rel)   (no relu)
    gemm_bf16_wmma<1, false, true><<<dim3(MT_AC, cdiv(H2, 64)), 256, 0, stream>>>(
        ha, Wc2, nullptr, nullptr, nullptr, nullptr,
        b2_tic, b2_rel, agg2t, agg2r, bufA, N_AC, H2, H1, 0.5f);  // ha2 in bufA
    // hw2 = hw@Ws2_w + agg2w + b2_w   -> second output
    gemm_bf16_wmma<1, false, true><<<dim3(MT_W, cdiv(H2, 64)), 256, 0, stream>>>(
        hw, Ws2_w, nullptr, nullptr, nullptr, nullptr,
        b2_w, nullptr, agg2w, nullptr, out_hw2, N_WD, H2, H1, 1.0f);

    // 7) MLP head on acoustic nodes
    gemm_bf16_wmma<1, true, true><<<dim3(MT_AC, 1), 256, 0, stream>>>(
        bufA, W_l1, nullptr, nullptr, nullptr, nullptr,
        b_l1, nullptr, nullptr, nullptr, agg2t, N_AC, LINH, H2, 1.0f);  // h1 (reuses agg2t)
    gemm_bf16_wmma<1, true, false><<<dim3(MT_AC, 1), 256, 0, stream>>>(
        agg2t, W_l2, nullptr, nullptr, nullptr, nullptr,
        b_l2, nullptr, nullptr, nullptr, agg2r, N_AC, 32, LINH, 1.0f);  // h2 (reuses agg2r)
    gemm_bf16_wmma<1, false, false><<<dim3(MT_AC, 1), 256, 0, stream>>>(
        agg2r, W_l3, nullptr, nullptr, nullptr, nullptr,
        b_l3, nullptr, nullptr, nullptr, out_head, N_AC, OUTD, 32, 1.0f);  // logits
}
